// AdaptiveMask_80221399155211
// MI455X (gfx1250) — matrix operand order repaired and checked by reading, not compiled, not run on hardware
//
#include <hip/hip_runtime.h>
#include <hip/hip_bf16.h>

// Problem constants (match reference).
#define Bn 32
#define Tn 1024
#define Dn 512
#define Hn 256
#define NLn 2
#define FOURH 1024
#define Mrows (Bn * Tn)   // 32768

typedef __attribute__((ext_vector_type(16))) __bf16 bf16x16;
typedef __attribute__((ext_vector_type(8)))  float  f32x8;

union Frag { uint4 q[2]; bf16x16 v; };

// Async global->LDS copy via CDNA5 GLOBAL_LOAD_ASYNC_TO_LDS_B128 (ASYNCcnt).
// The clang builtin exists but demands cuda_device/cuda_shared qualified
// pointer types that HIP source cannot spell; inline asm sidesteps that.
#define USE_ASYNC_ASM 1

__device__ __forceinline__ void async_g2l_b128(const void* gptr, void* lptr) {
  unsigned lds = (unsigned)(size_t)lptr;              // LDS address is the pointer's low 32 bits
  unsigned long long ga = (unsigned long long)(size_t)gptr;
  asm volatile("global_load_async_to_lds_b128 %0, %1, off"
               :: "v"(lds), "v"(ga) : "memory");
}
__device__ __forceinline__ void wait_async0() {
  asm volatile("s_wait_asynccnt 0x0" ::: "memory");
}

// ---------------------------------------------------------------------------
// f32 -> bf16 (linear)
// ---------------------------------------------------------------------------
__global__ void cvt_f32_bf16(const float* __restrict__ src,
                             __bf16* __restrict__ dst, int n) {
  int i = blockIdx.x * blockDim.x + threadIdx.x;
  if (i < n) dst[i] = (__bf16)src[i];
}

// biasC = bih + bhh  (per layer/dir/gate-col)
__global__ void bias_sum(const float* __restrict__ bih,
                         const float* __restrict__ bhh,
                         float* __restrict__ bc, int n) {
  int i = blockIdx.x * blockDim.x + threadIdx.x;
  if (i < n) bc[i] = bih[i] + bhh[i];
}

// ---------------------------------------------------------------------------
// Pack Whh (f32 [l][d][1024][256]) into fragment-linear bf16 so the scan
// kernel can load WMMA B-fragments with coalesced 128-bit loads.
// Per (l,d): 64 n-tiles x 8 k-tiles; fragment = 512 bf16 in lane order:
//   lane L (col = L&15, h = L>>4) holds K = 8h+0..7 then K = 16+8h+0..7.
// ---------------------------------------------------------------------------
__global__ void pack_whh(const float* __restrict__ Whh,
                         __bf16* __restrict__ out, int total) {
  int e = blockIdx.x * blockDim.x + threadIdx.x;
  if (e >= total) return;
  int ld   = e >> 18;        // 1024*256 = 262144 per (layer,dir)
  int r    = e & 262143;
  int frag = r >> 9;         // 512 elems per fragment
  int off  = r & 511;
  int lane = off >> 4, j = off & 15;
  int nt = frag >> 3, kt = frag & 7;
  int nn = nt * 16 + (lane & 15);
  int kk = kt * 32 + ((j >> 3) << 4) + ((lane >> 4) << 3) + (j & 7);
  out[e] = (__bf16)Whh[(size_t)ld * 262144 + (size_t)nn * Hn + kk];
}

// ---------------------------------------------------------------------------
// Pre-gate GEMM:  G[m][n] = sum_k X[m][k] * W[n][k] + bias[n]
// X: [32768][512] bf16 row-major, W: [1024][512] bf16 row-major.
// WG = 256 threads (8 waves), block tile 128M x 64N, K-steps of 32.
// Double-buffered LDS staging; async global->LDS copies.
// ---------------------------------------------------------------------------
#define GM_BLK 128
#define GN_BLK 64
#define TILE_BYTES ((GM_BLK + GN_BLK) * 80)  // row pitch 80B (64 data + pad)

__global__ __launch_bounds__(256, 1)
void gemm_pregate(const __bf16* __restrict__ X,
                  const __bf16* __restrict__ W,
                  const float*  __restrict__ bias,
                  float* __restrict__ G) {
  __shared__ __align__(16) unsigned char sm[2 * TILE_BYTES];

  const int tid = threadIdx.x;
  const int lane = tid & 31, w = tid >> 5;
  const int m0 = blockIdx.y * GM_BLK;
  const int n0 = blockIdx.x * GN_BLK;
  const int wm = w >> 1, wn = w & 1;      // 4x2 wave grid, wave tile 32x32

  // per-thread staging coordinates (three 16B chunks per thread per k-step)
  const int ar0 = tid >> 2,        ac = tid & 3;   // A rows 0..63
  const int ar1 = (tid >> 2) + 64;                 // A rows 64..127
  const int br  = tid >> 2;                        // B rows 0..63
  const __bf16* gA0 = X + (size_t)(m0 + ar0) * Dn + ac * 8;
  const __bf16* gA1 = X + (size_t)(m0 + ar1) * Dn + ac * 8;
  const __bf16* gB  = W + (size_t)(n0 + br)  * Dn + ac * 8;
  const int lA0 = ar0 * 80 + ac * 16;
  const int lA1 = ar1 * 80 + ac * 16;
  const int lB  = GM_BLK * 80 + br * 80 + ac * 16;

  f32x8 acc[2][2];
  for (int i2 = 0; i2 < 2; i2++)
    for (int j2 = 0; j2 < 2; j2++) {
      int nn = n0 + (wn * 2 + j2) * 16 + (lane & 15);
      float bv = bias[nn];
      for (int v = 0; v < 8; v++) acc[i2][j2][v] = bv;
    }

  // ---- prime buffer 0 with k-step 0 ----
#if USE_ASYNC_ASM
  async_g2l_b128(gA0, sm + lA0);
  async_g2l_b128(gA1, sm + lA1);
  async_g2l_b128(gB,  sm + lB);
  wait_async0();
#else
  {
    uint4 r0 = *(const uint4*)gA0;
    uint4 r1 = *(const uint4*)gA1;
    uint4 r2 = *(const uint4*)gB;
    *(uint4*)(sm + lA0) = r0;
    *(uint4*)(sm + lA1) = r1;
    *(uint4*)(sm + lB)  = r2;
  }
#endif
  __syncthreads();

  for (int ki = 0; ki < 16; ki++) {
    unsigned char* cur = sm + (ki & 1) * TILE_BYTES;
    unsigned char* nxt = sm + ((ki + 1) & 1) * TILE_BYTES;
    const int kk = (ki + 1) * 32;
    const bool more = ki < 15;

    // ---- issue next tile's copy while we compute on the current one ----
#if USE_ASYNC_ASM
    if (more) {
      async_g2l_b128(gA0 + kk, nxt + lA0);
      async_g2l_b128(gA1 + kk, nxt + lA1);
      async_g2l_b128(gB + kk,  nxt + lB);
    }
#else
    uint4 r0, r1, r2;
    if (more) {
      r0 = *(const uint4*)(gA0 + kk);
      r1 = *(const uint4*)(gA1 + kk);
      r2 = *(const uint4*)(gB + kk);
      if (ki + 2 < 16) {
        __builtin_prefetch(gA0 + kk + 32, 0, 1);
        __builtin_prefetch(gA1 + kk + 32, 0, 1);
        __builtin_prefetch(gB + kk + 32, 0, 1);
      }
    }
#endif

    // ---- 4 WMMAs per wave on the current tile ----
    unsigned char* At = cur;
    unsigned char* Bt = cur + GM_BLK * 80;
    for (int i2 = 0; i2 < 2; i2++) {
      Frag a;
      int mm = (wm * 2 + i2) * 16 + (lane & 15);
      const unsigned char* ap = At + mm * 80 + ((lane >> 4) << 4);
      a.q[0] = *(const uint4*)ap;
      a.q[1] = *(const uint4*)(ap + 32);
      for (int j2 = 0; j2 < 2; j2++) {
        Frag bfr;
        int nn = (wn * 2 + j2) * 16 + (lane & 15);
        const unsigned char* bp = Bt + nn * 80 + ((lane >> 4) << 4);
        bfr.q[0] = *(const uint4*)bp;
        bfr.q[1] = *(const uint4*)(bp + 32);
        acc[i2][j2] = __builtin_amdgcn_wmma_f32_16x16x32_bf16(
            false, a.v, false, bfr.v, (short)0, acc[i2][j2], false, false);
      }
    }

#if USE_ASYNC_ASM
    if (more) wait_async0();
#else
    if (more) {
      *(uint4*)(nxt + lA0) = r0;
      *(uint4*)(nxt + lA1) = r1;
      *(uint4*)(nxt + lB)  = r2;
    }
#endif
    __syncthreads();
  }

  for (int i2 = 0; i2 < 2; i2++)
    for (int j2 = 0; j2 < 2; j2++) {
      int nn = n0 + (wn * 2 + j2) * 16 + (lane & 15);
      for (int v = 0; v < 8; v++) {
        int mm = m0 + (wm * 2 + i2) * 16 + v + ((lane >> 4) << 3);
        G[(size_t)mm * FOURH + nn] = acc[i2][j2][v];
      }
    }
}

// ---------------------------------------------------------------------------
// Recurrent scan. grid.x = direction (0 fwd, 1 bwd). 1024 threads = 32 waves.
// Wave w: mt = w>>4 (batch tile), c = w&15 (hidden-col tile). Each wave owns
// i/f/g/o gate tiles for its (16 batch x 16 hidden) block -> fully in-register
// elementwise LSTM cell, c-state persistent in VGPRs, h-state bf16 in LDS.
// ---------------------------------------------------------------------------
__global__ __launch_bounds__(1024, 1)
void lstm_scan(const float* __restrict__ G,      // [2][Mrows][1024] pre-gates
               const __bf16* __restrict__ WhhP,  // [2][512 frags][512] packed
               const int* __restrict__ lens,     // [B]
               float* __restrict__ out) {        // [B][T][512]
  const int dir  = blockIdx.x;
  const int tid  = threadIdx.x;
  const int lane = tid & 31, w = tid >> 5;
  const int c  = w & 15;     // hidden-col tile (16 cols)
  const int mt = w >> 4;     // batch tile (16 rows)

  __shared__ __align__(16) __bf16 hbf[32][264];  // pitch 528B (16B multiple)
  __shared__ int slen[32];

  for (int i = tid; i < 32 * 264; i += 1024) (&hbf[0][0])[i] = (__bf16)0.f;
  if (tid < 32) slen[tid] = lens[tid];
  __syncthreads();

  const float*  Gd = G + (size_t)dir * Mrows * FOURH;
  const __bf16* Wd = WhhP + (size_t)dir * 512 * 512;

  f32x8 creg;
  for (int v = 0; v < 8; v++) creg[v] = 0.f;

  int bidx[8];
  for (int v = 0; v < 8; v++) bidx[v] = mt * 16 + v + ((lane >> 4) << 3);
  const int hc = c * 16 + (lane & 15);

  for (int t = 0; t < Tn; t++) {
    int trow[8];
    bool act[8];
    for (int v = 0; v < 8; v++) {
      int L = slen[bidx[v]];
      act[v]  = t < L;
      trow[v] = (dir == 0) ? t : (act[v] ? (L - 1 - t) : 0);
    }
    // init accumulators with pre-gates (bias already folded in)
    f32x8 acc[4];
    for (int g = 0; g < 4; g++) {
      int nn = g * 256 + hc;
      for (int v = 0; v < 8; v++)
        acc[g][v] = Gd[((size_t)bidx[v] * Tn + trow[v]) * FOURH + nn];
    }
    // prefetch next step's pre-gate rows while the WMMAs run
    if (t + 1 < Tn) {
      for (int v = 0; v < 8; v++) {
        int L = slen[bidx[v]];
        int tnx = (dir == 0) ? (t + 1) : ((t + 1 < L) ? (L - 2 - t) : 0);
        const float* p = &Gd[((size_t)bidx[v] * Tn + tnx) * FOURH + hc];
        __builtin_prefetch(p, 0, 1);
        __builtin_prefetch(p + 512, 0, 1);   // covers g/o gate half as well
      }
    }
    // gates += h @ Whh^T  (K = 256 in 8 WMMA k-steps)
    for (int kt = 0; kt < 8; kt++) {
      Frag a;
      const unsigned char* ap = (const unsigned char*)&hbf[0][0] +
          (mt * 16 + (lane & 15)) * 528 + kt * 64 + ((lane >> 4) << 4);
      a.q[0] = *(const uint4*)ap;
      a.q[1] = *(const uint4*)(ap + 32);
      for (int g = 0; g < 4; g++) {
        Frag bb;
        int frag = (g * 16 + c) * 8 + kt;
        const uint4* bp = (const uint4*)(Wd + (size_t)frag * 512) + lane * 2;
        bb.q[0] = bp[0];
        bb.q[1] = bp[1];
        acc[g] = __builtin_amdgcn_wmma_f32_16x16x32_bf16(
            false, a.v, false, bb.v, (short)0, acc[g], false, false);
      }
    }
    __syncthreads();  // all waves done reading hbf

    for (int v = 0; v < 8; v++) {
      float ig = 1.f / (1.f + __expf(-acc[0][v]));
      float fg = 1.f / (1.f + __expf(-acc[1][v]));
      float gg = tanhf(acc[2][v]);
      float og = 1.f / (1.f + __expf(-acc[3][v]));
      float cn = fg * creg[v] + ig * gg;
      float hn = og * tanhf(cn);
      int b = bidx[v];
      if (act[v]) {
        creg[v] = cn;
        hbf[b][hc] = (__bf16)hn;
        int to = (dir == 0) ? t : trow[v];
        out[((size_t)b * Tn + to) * Dn + dir * Hn + hc] = hn;
      } else {
        out[((size_t)b * Tn + t) * Dn + dir * Hn + hc] = 0.f;
      }
    }
    __syncthreads();  // hbf updated for next step
  }
}

// ---------------------------------------------------------------------------
// logits[r] = dot(lstm_o[r, :512], linW) + linb   (wave per row)
// ---------------------------------------------------------------------------
__global__ void logits_kernel(const float* __restrict__ ho,
                              const float* __restrict__ lw,
                              const float* __restrict__ lb,
                              float* __restrict__ logits) {
  int r = blockIdx.x * 8 + (threadIdx.x >> 5);
  int lane = threadIdx.x & 31;
  const float* row = ho + (size_t)r * Dn;
  float s = 0.f;
  for (int i = lane; i < Dn; i += 32) s += row[i] * lw[i];
  for (int o = 16; o > 0; o >>= 1) s += __shfl_xor(s, o, 32);
  if (lane == 0) logits[r] = s + lb[0];
}

// ---------------------------------------------------------------------------
// Two smallest logits per batch (softmax is monotone -> top_k(-probs, 2)).
// Lexicographic (value, index) to mimic top_k tie-breaking.
// ---------------------------------------------------------------------------
struct Best2 { float v0, v1; int i0, i1; };
__device__ inline void best2_insert(Best2& b, float v, int i) {
  if (v < b.v0 || (v == b.v0 && i < b.i0)) {
    b.v1 = b.v0; b.i1 = b.i0; b.v0 = v; b.i0 = i;
  } else if (v < b.v1 || (v == b.v1 && i < b.i1)) {
    b.v1 = v; b.i1 = i;
  }
}
__global__ void argmin2_kernel(const float* __restrict__ logits,
                               int* __restrict__ idx) {
  __shared__ float sv[512];
  __shared__ int   si[512];
  int b = blockIdx.x, tid = threadIdx.x;
  Best2 best{3.4e38f, 3.4e38f, 0x7fffffff, 0x7fffffff};
  for (int t = tid; t < Tn; t += 256) best2_insert(best, logits[b * Tn + t], t);
  sv[tid * 2] = best.v0; sv[tid * 2 + 1] = best.v1;
  si[tid * 2] = best.i0; si[tid * 2 + 1] = best.i1;
  __syncthreads();
  for (int s = 128; s > 0; s >>= 1) {
    if (tid < s) {
      Best2 m{sv[tid * 2], sv[tid * 2 + 1], si[tid * 2], si[tid * 2 + 1]};
      best2_insert(m, sv[(tid + s) * 2],     si[(tid + s) * 2]);
      best2_insert(m, sv[(tid + s) * 2 + 1], si[(tid + s) * 2 + 1]);
      sv[tid * 2] = m.v0; sv[tid * 2 + 1] = m.v1;
      si[tid * 2] = m.i0; si[tid * 2 + 1] = m.i1;
    }
    __syncthreads();
  }
  if (tid == 0) { idx[b * 2] = si[0]; idx[b * 2 + 1] = si[1]; }
}

// ---------------------------------------------------------------------------
// LayerNorm over the 512 features (wave per row).
// ---------------------------------------------------------------------------
__global__ void layernorm_kernel(const float* __restrict__ ho,
                                 const float* __restrict__ gamma,
                                 const float* __restrict__ beta,
                                 float* __restrict__ outp) {
  int r = blockIdx.x * 8 + (threadIdx.x >> 5);
  int lane = threadIdx.x & 31;
  const float* row = ho + (size_t)r * Dn;
  float s = 0.f, s2 = 0.f;
  for (int i = lane; i < Dn; i += 32) {
    float x = row[i];
    s += x; s2 += x * x;
  }
  for (int o = 16; o > 0; o >>= 1) {
    s  += __shfl_xor(s, o, 32);
    s2 += __shfl_xor(s2, o, 32);
  }
  float mu  = s * (1.f / Dn);
  float var = s2 * (1.f / Dn) - mu * mu;
  float inv = rsqrtf(var + 1e-5f);
  for (int i = lane; i < Dn; i += 32)
    outp[(size_t)r * Dn + i] = (row[i] - mu) * inv * gamma[i] + beta[i];
}

__global__ void mask_kernel(const unsigned char* __restrict__ mask,
                            const int* __restrict__ idx,
                            const int* __restrict__ kptr,
                            float* __restrict__ outm) {
  int r = blockIdx.x * blockDim.x + threadIdx.x;
  if (r >= Bn * Tn) return;
  int b = r >> 10, t = r & 1023;
  int k = kptr[0];
  bool m = mask[r] != 0;
  if (k >= 1 && t == idx[b * 2])     m = false;
  if (k >= 2 && t == idx[b * 2 + 1]) m = false;
  outm[r] = m ? 1.f : 0.f;
}

// ---------------------------------------------------------------------------
extern "C" void kernel_launch(void* const* d_in, const int* in_sizes, int n_in,
                              void* d_out, int out_size, void* d_ws,
                              size_t ws_size, hipStream_t stream) {
  const float* x     = (const float*)d_in[0];
  const int*   lens  = (const int*)d_in[1];
  const int*   kptr  = (const int*)d_in[2];
  const unsigned char* mask = (const unsigned char*)d_in[3];
  const float* Wih   = (const float*)d_in[4];
  const float* Whh   = (const float*)d_in[5];
  const float* bih   = (const float*)d_in[6];
  const float* bhh   = (const float*)d_in[7];
  const float* linW  = (const float*)d_in[8];
  const float* linb  = (const float*)d_in[9];
  const float* gamma = (const float*)d_in[10];
  const float* beta  = (const float*)d_in[11];
  float* out = (float*)d_out;

  // workspace carve-up (all offsets 16B-aligned)
  char* ws = (char*)d_ws;
  size_t off = 0;
  __bf16* WihB = (__bf16*)(ws + off); off += (size_t)NLn * 2 * FOURH * Dn * 2; // 4 MB
  __bf16* WhhP = (__bf16*)(ws + off); off += (size_t)NLn * 2 * FOURH * Hn * 2; // 2 MB
  float*  biasC = (float*)(ws + off); off += (size_t)NLn * 2 * FOURH * 4;      // 16 KB
  __bf16* Xbf  = (__bf16*)(ws + off); off += (size_t)Mrows * Dn * 2;           // 32 MB
  float*  G    = (float*)(ws + off);  off += (size_t)2 * Mrows * FOURH * 4;    // 256 MB
  float*  hbuf = (float*)(ws + off);  off += (size_t)Mrows * Dn * 4;           // 64 MB
  float*  logits = (float*)(ws + off); off += (size_t)Mrows * 4;               // 128 KB
  int*    idx  = (int*)(ws + off);    off += (size_t)Bn * 2 * 4;

  // weight preparation
  {
    int nWih = NLn * 2 * FOURH * Dn;      // 2,097,152
    cvt_f32_bf16<<<nWih / 256, 256, 0, stream>>>(Wih, WihB, nWih);
    int nWhh = NLn * 2 * FOURH * Hn;      // 1,048,576
    pack_whh<<<nWhh / 256, 256, 0, stream>>>(Whh, WhhP, nWhh);
    int nB = NLn * 2 * FOURH;
    bias_sum<<<nB / 256, 256, 0, stream>>>(bih, bhh, biasC, nB);
  }

  // layer-0 input -> bf16
  int nX = Mrows * Dn;                    // 16,777,216
  cvt_f32_bf16<<<nX / 256, 256, 0, stream>>>(x, Xbf, nX);

  for (int layer = 0; layer < NLn; layer++) {
    for (int dir = 0; dir < 2; dir++) {
      gemm_pregate<<<dim3(FOURH / GN_BLK, Mrows / GM_BLK), 256, 0, stream>>>(
          Xbf,
          WihB + (size_t)(layer * 2 + dir) * FOURH * Dn,
          biasC + (size_t)(layer * 2 + dir) * FOURH,
          G + (size_t)dir * Mrows * FOURH);
    }
    lstm_scan<<<2, 1024, 0, stream>>>(
        G, WhhP + (size_t)layer * 2 * 512 * 512, lens, hbuf);
    if (layer + 1 < NLn)
      cvt_f32_bf16<<<nX / 256, 256, 0, stream>>>(hbuf, Xbf, nX);
  }

  // post-processing
  logits_kernel<<<Mrows / 8 / 32, 256, 0, stream>>>(hbuf, linW, linb, logits);
  argmin2_kernel<<<Bn, 256, 0, stream>>>(logits, idx);
  layernorm_kernel<<<Mrows / 8 / 32, 256, 0, stream>>>(hbuf, gamma, beta, out);
  mask_kernel<<<(Bn * Tn) / 256, 256, 0, stream>>>(
      mask, idx, kptr, out + (size_t)Mrows * Dn);
}
